// GraphNodeEncoder_17772574671465
// MI455X (gfx1250) — compile-verified, matrix operand order checked by reading, not run on hardware
//
#include <hip/hip_runtime.h>

typedef __bf16 bf16_t;
typedef __attribute__((ext_vector_type(16))) __bf16 v16bf;
typedef __attribute__((ext_vector_type(8)))  __bf16 v8bf;
typedef __attribute__((ext_vector_type(8)))  float  v8f;

// ---------- helpers ----------
static __device__ __forceinline__ unsigned f32_key(float f) {
  unsigned u = __float_as_uint(f);
  return (u & 0x80000000u) ? ~u : (u | 0x80000000u);
}
static __device__ __forceinline__ float key_f32(unsigned k) {
  unsigned u = (k & 0x80000000u) ? (k & 0x7fffffffu) : ~k;
  return __uint_as_float(u);
}
static __device__ __forceinline__ void edge_endpoints(const int* ei, int E, int e,
                                                      int& s, int& d) {
  if (e < E) { s = ei[e]; d = ei[E + e]; }
  else       { s = e - E; d = e - E; }      // self-loops appended after real edges
}

// ---------- small utility kernels ----------
__global__ void fill_u32_kernel(unsigned* __restrict__ p, unsigned v, long long n) {
  long long t = (long long)blockIdx.x * blockDim.x + threadIdx.x;
  if (t < n) p[t] = v;
}

// W [K x Nw] f32 row-major  ->  Wt [Nw x K] bf16 (so GEMM B-columns are contiguous)
__global__ void transpose_cvt_kernel(const float* __restrict__ W, bf16_t* __restrict__ Wt,
                                     int K, int Nw) {
  int t = blockIdx.x * blockDim.x + threadIdx.x;
  if (t >= K * Nw) return;
  int k = t / Nw, n = t - k * Nw;
  Wt[(size_t)n * K + k] = (bf16_t)W[t];
}

// h0[n, c] = (bf16) emb[x[n], c]
__global__ void gather_bf16_kernel(const int* __restrict__ x, const float* __restrict__ emb,
                                   bf16_t* __restrict__ h0, int Nn, int D) {
  long long t = (long long)blockIdx.x * blockDim.x + threadIdx.x;
  if (t >= (long long)Nn * D) return;
  int n = (int)(t / D), c = (int)(t - (long long)n * D);
  h0[t] = (bf16_t)emb[(size_t)x[n] * D + c];
}

// ---------- WMMA GEMM: C[MxNw] = A[MxK] * Bt[NwxK]^T (+bias per column) ----------
// One wave -> one 16x16 f32 tile. K multiple of 32, M,Nw multiples of 16.
__global__ void gemm_bf16_wmma_kernel(const bf16_t* __restrict__ A,
                                      const bf16_t* __restrict__ Bt,
                                      float* __restrict__ C,
                                      const float* __restrict__ bias,
                                      int M, int K, int Nw) {
  int gw   = (blockIdx.x * blockDim.x + threadIdx.x) >> 5;
  int lane = threadIdx.x & 31;
  int tn_count = Nw >> 4;
  int ntiles   = (M >> 4) * tn_count;
  if (gw >= ntiles) return;                 // wave-uniform: EXEC all-ones inside
  int tm = gw / tn_count;
  int tn = gw - tm * tn_count;

  int l15   = lane & 15;
  int khalf = (lane >> 4) << 3;             // 0 or 8

  const bf16_t* arow = A  + (size_t)((tm << 4) + l15) * K + khalf;
  const bf16_t* brow = Bt + (size_t)((tn << 4) + l15) * K + khalf;

  v8f acc;
#pragma unroll
  for (int i = 0; i < 8; ++i) acc[i] = 0.0f;

  for (int kk = 0; kk < K; kk += 32) {
    union { v16bf v; v8bf h[2]; } ua, ub;
    ua.h[0] = *(const v8bf*)(arow + kk);        // K = kk+khalf+0..7
    ua.h[1] = *(const v8bf*)(arow + kk + 16);   // K = kk+16+khalf+0..7
    ub.h[0] = *(const v8bf*)(brow + kk);
    ub.h[1] = *(const v8bf*)(brow + kk + 16);
    acc = __builtin_amdgcn_wmma_f32_16x16x32_bf16(false, ua.v, false, ub.v,
                                                  (short)0, acc, false, false);
  }

  int col  = (tn << 4) + l15;
  int row0 = (tm << 4) + ((lane >> 4) << 3);
  float bv = bias ? bias[col] : 0.0f;
#pragma unroll
  for (int r = 0; r < 8; ++r)
    C[(size_t)(row0 + r) * Nw + col] = acc[r] + bv;
}

// ---------- attention coefficients: als/ald[n,h] = <feat[n,h,:], a_src/dst[h,:]> ----------
__global__ void attn_coef_kernel(const float* __restrict__ feat,
                                 const float* __restrict__ a_src,
                                 const float* __restrict__ a_dst,
                                 float* __restrict__ als, float* __restrict__ ald,
                                 int Nn, int H, int Cc) {
  long long t = (long long)blockIdx.x * blockDim.x + threadIdx.x;
  if (t >= (long long)Nn * H) return;
  int n = (int)(t / H), h = (int)(t - (long long)n * H);
  const float* f  = feat  + (size_t)n * H * Cc + (size_t)h * Cc;
  const float* as = a_src + (size_t)h * Cc;
  const float* ad = a_dst + (size_t)h * Cc;
  float s1 = 0.f, s2 = 0.f;
  for (int c = 0; c < Cc; ++c) { float v = f[c]; s1 += v * as[c]; s2 += v * ad[c]; }
  als[t] = s1; ald[t] = s2;
}

// ---------- segment softmax pieces ----------
__global__ void edge_max_kernel(const int* __restrict__ ei, int E, int Nn, int H,
                                const float* __restrict__ als, const float* __restrict__ ald,
                                unsigned* __restrict__ mx) {
  long long t = (long long)blockIdx.x * blockDim.x + threadIdx.x;
  long long total = (long long)(E + Nn) * H;
  if (t >= total) return;
  int e = (int)(t / H), h = (int)(t - (long long)e * H);
  int s, d; edge_endpoints(ei, E, e, s, d);
  float v = als[(size_t)s * H + h] + ald[(size_t)d * H + h];
  v = v > 0.f ? v : 0.2f * v;                           // leaky_relu(0.2)
  atomicMax(&mx[(size_t)d * H + h], f32_key(v));
}

__global__ void edge_exp_kernel(const int* __restrict__ ei, int E, int Nn, int H,
                                const float* __restrict__ als, const float* __restrict__ ald,
                                const unsigned* __restrict__ mx,
                                float* __restrict__ exbuf, float* __restrict__ sm) {
  long long t = (long long)blockIdx.x * blockDim.x + threadIdx.x;
  long long total = (long long)(E + Nn) * H;
  if (t >= total) return;
  int e = (int)(t / H), h = (int)(t - (long long)e * H);
  int s, d; edge_endpoints(ei, E, e, s, d);
  float v = als[(size_t)s * H + h] + ald[(size_t)d * H + h];
  v = v > 0.f ? v : 0.2f * v;
  float m  = key_f32(mx[(size_t)d * H + h]);
  float ex = expf(v - m);
  exbuf[(size_t)e * H + h] = ex;
  atomicAdd(&sm[(size_t)d * H + h], ex);
}

// one block per edge, blockDim = H*Cc (channel-contiguous -> coalesced gather+scatter)
__global__ void edge_agg_kernel(const int* __restrict__ ei, int E, int Nn, int H, int Cc,
                                const float* __restrict__ feat,
                                const float* __restrict__ exbuf,
                                const float* __restrict__ sm,
                                float* __restrict__ out) {
  int e = blockIdx.x;
  int c = threadIdx.x;
  int HD = H * Cc;
  int s, d; edge_endpoints(ei, E, e, s, d);
  int h = c / Cc;
  float alpha = exbuf[(size_t)e * H + h] / (sm[(size_t)d * H + h] + 1e-16f);
  atomicAdd(&out[(size_t)d * HD + c], feat[(size_t)s * HD + c] * alpha);
}

// inb[n,c] = (bf16)(acc[n,c] + b[c])
__global__ void bias_cvt_kernel(const float* __restrict__ acc, const float* __restrict__ b,
                                bf16_t* __restrict__ inb, int Nn, int HD) {
  long long t = (long long)blockIdx.x * blockDim.x + threadIdx.x;
  if (t >= (long long)Nn * HD) return;
  int c = (int)(t % HD);
  inb[t] = (bf16_t)(acc[t] + b[c]);
}

// ---------- host ----------
static inline int cdiv_ll(long long a, long long b) { return (int)((a + b - 1) / b); }

extern "C" void kernel_launch(void* const* d_in, const int* in_sizes, int n_in,
                              void* d_out, int out_size, void* d_ws, size_t ws_size,
                              hipStream_t stream) {
  const int*   x    = (const int*)  d_in[0];
  const int*   ei   = (const int*)  d_in[1];
  const float* emb  = (const float*)d_in[2];
  const float* W1   = (const float*)d_in[3];
  const float* as1  = (const float*)d_in[4];
  const float* ad1  = (const float*)d_in[5];
  const float* b1   = (const float*)d_in[6];
  const float* W2   = (const float*)d_in[7];
  const float* as2  = (const float*)d_in[8];
  const float* ad2  = (const float*)d_in[9];
  const float* b2   = (const float*)d_in[10];
  const float* Wo   = (const float*)d_in[11];
  const float* bo   = (const float*)d_in[12];
  float* out        = (float*)d_out;

  const int Nn  = in_sizes[0];
  const int E   = in_sizes[1] / 2;
  const int D   = in_sizes[2] / Nn;        // 128
  const int HD1 = in_sizes[3] / D;         // 256
  const int H1  = 8, C1 = HD1 / 8;         // 8 heads x 32
  const int OD  = in_sizes[12];            // 128
  const int Et  = E + Nn;

  // bump allocator over workspace
  char*  base = (char*)d_ws;
  size_t off  = 0;
  auto take = [&](size_t nbytes) -> void* {
    void* r = base + off;
    off += (nbytes + 255) & ~(size_t)255;
    return r;
  };

  // persistent
  bf16_t* wt1  = (bf16_t*)take((size_t)HD1 * D * 2);   // [256 x 128]
  bf16_t* wt2  = (bf16_t*)take((size_t)OD * HD1 * 2);  // [128 x 256]
  bf16_t* wto  = (bf16_t*)take((size_t)OD * OD * 2);   // [128 x 128]
  bf16_t* in1b = (bf16_t*)take((size_t)Nn * HD1 * 2);
  bf16_t* in2b = (bf16_t*)take((size_t)Nn * OD * 2);
  size_t scratch0 = off;

  // ---- phase 1 scratch ----
  bf16_t*   h0b   = (bf16_t*)  take((size_t)Nn * D * 2);
  float*    feat1 = (float*)   take((size_t)Nn * HD1 * 4);
  float*    als1  = (float*)   take((size_t)Nn * H1 * 4);
  float*    ald1  = (float*)   take((size_t)Nn * H1 * 4);
  unsigned* mx1   = (unsigned*)take((size_t)Nn * H1 * 4);
  float*    sm1   = (float*)   take((size_t)Nn * H1 * 4);
  float*    ex1   = (float*)   take((size_t)Et * H1 * 4);
  float*    out1  = (float*)   take((size_t)Nn * HD1 * 4);

  const int B = 256;

  // weight prep
  transpose_cvt_kernel<<<cdiv_ll((long long)D * HD1, B), B, 0, stream>>>(W1, wt1, D, HD1);
  transpose_cvt_kernel<<<cdiv_ll((long long)HD1 * OD, B), B, 0, stream>>>(W2, wt2, HD1, OD);
  transpose_cvt_kernel<<<cdiv_ll((long long)OD * OD, B), B, 0, stream>>>(Wo, wto, OD, OD);

  // conv1
  gather_bf16_kernel<<<cdiv_ll((long long)Nn * D, B), B, 0, stream>>>(x, emb, h0b, Nn, D);
  {
    long long ntiles = (long long)(Nn / 16) * (HD1 / 16);
    gemm_bf16_wmma_kernel<<<cdiv_ll(ntiles, 8), B, 0, stream>>>(h0b, wt1, feat1, nullptr,
                                                                Nn, D, HD1);
  }
  attn_coef_kernel<<<cdiv_ll((long long)Nn * H1, B), B, 0, stream>>>(feat1, as1, ad1,
                                                                     als1, ald1, Nn, H1, C1);
  fill_u32_kernel<<<cdiv_ll((long long)Nn * H1, B), B, 0, stream>>>(mx1, 0u, (long long)Nn * H1);
  fill_u32_kernel<<<cdiv_ll((long long)Nn * H1, B), B, 0, stream>>>((unsigned*)sm1, 0u,
                                                                    (long long)Nn * H1);
  fill_u32_kernel<<<cdiv_ll((long long)Nn * HD1, B), B, 0, stream>>>((unsigned*)out1, 0u,
                                                                     (long long)Nn * HD1);
  edge_max_kernel<<<cdiv_ll((long long)Et * H1, B), B, 0, stream>>>(ei, E, Nn, H1,
                                                                    als1, ald1, mx1);
  edge_exp_kernel<<<cdiv_ll((long long)Et * H1, B), B, 0, stream>>>(ei, E, Nn, H1,
                                                                    als1, ald1, mx1, ex1, sm1);
  edge_agg_kernel<<<Et, HD1, 0, stream>>>(ei, E, Nn, H1, C1, feat1, ex1, sm1, out1);
  bias_cvt_kernel<<<cdiv_ll((long long)Nn * HD1, B), B, 0, stream>>>(out1, b1, in1b, Nn, HD1);

  // ---- phase 2 scratch (reuse phase-1 region) ----
  off = scratch0;
  float*    feat2 = (float*)   take((size_t)Nn * OD * 4);
  float*    als2  = (float*)   take((size_t)Nn * 4);
  float*    ald2  = (float*)   take((size_t)Nn * 4);
  unsigned* mx2   = (unsigned*)take((size_t)Nn * 4);
  float*    sm2   = (float*)   take((size_t)Nn * 4);
  float*    ex2   = (float*)   take((size_t)Et * 4);
  float*    out2  = (float*)   take((size_t)Nn * OD * 4);

  // conv2 (1 head, concat=False with 1 head == identity reduction)
  {
    long long ntiles = (long long)(Nn / 16) * (OD / 16);
    gemm_bf16_wmma_kernel<<<cdiv_ll(ntiles, 8), B, 0, stream>>>(in1b, wt2, feat2, nullptr,
                                                                Nn, HD1, OD);
  }
  attn_coef_kernel<<<cdiv_ll((long long)Nn, B), B, 0, stream>>>(feat2, as2, ad2,
                                                                als2, ald2, Nn, 1, OD);
  fill_u32_kernel<<<cdiv_ll((long long)Nn, B), B, 0, stream>>>(mx2, 0u, (long long)Nn);
  fill_u32_kernel<<<cdiv_ll((long long)Nn, B), B, 0, stream>>>((unsigned*)sm2, 0u, (long long)Nn);
  fill_u32_kernel<<<cdiv_ll((long long)Nn * OD, B), B, 0, stream>>>((unsigned*)out2, 0u,
                                                                    (long long)Nn * OD);
  edge_max_kernel<<<cdiv_ll((long long)Et, B), B, 0, stream>>>(ei, E, Nn, 1, als2, ald2, mx2);
  edge_exp_kernel<<<cdiv_ll((long long)Et, B), B, 0, stream>>>(ei, E, Nn, 1,
                                                               als2, ald2, mx2, ex2, sm2);
  edge_agg_kernel<<<Et, OD, 0, stream>>>(ei, E, Nn, 1, OD, feat2, ex2, sm2, out2);
  bias_cvt_kernel<<<cdiv_ll((long long)Nn * OD, B), B, 0, stream>>>(out2, b2, in2b, Nn, OD);

  // final linear: out = in2 @ Wo + bo
  {
    long long ntiles = (long long)(Nn / 16) * (OD / 16);
    gemm_bf16_wmma_kernel<<<cdiv_ll(ntiles, 8), B, 0, stream>>>(in2b, wto, out, bo,
                                                                Nn, OD, OD);
  }
  (void)n_in; (void)out_size; (void)ws_size;
}
